// ActorMpnn_12446815223931
// MI455X (gfx1250) — matrix operand order repaired and checked by reading, not compiled, not run on hardware
//
#include <hip/hip_runtime.h>

// ---------------- types ----------------
typedef __attribute__((ext_vector_type(16))) __bf16 v16bf;
typedef __attribute__((ext_vector_type(8)))  __bf16 v8bf;
typedef __attribute__((ext_vector_type(8)))  float  v8f;

union ABfrag { v16bf v; v8bf h[2]; };

#define BATCH 4
#define NNODE 8192
#define NEDGE 131072
#define NODEF 16
#define HID   128

// ---------------- helpers ----------------
__device__ __forceinline__ unsigned enc_f32(float f) {
    unsigned b = __float_as_uint(f);
    return (b & 0x80000000u) ? ~b : (b | 0x80000000u);
}
__device__ __forceinline__ float dec_f32(unsigned e) {
    unsigned b = (e & 0x80000000u) ? (e & 0x7FFFFFFFu) : ~e;
    return __uint_as_float(b);
}
__device__ __forceinline__ float lrelu(float v) { return v > 0.f ? v : 0.01f * v; }

// store 4 floats as 4 bf16 (compiler packs into cvt_pk + ds_store_b64)
__device__ __forceinline__ void st4bf(__bf16* p, float4 v) {
    p[0] = (__bf16)v.x; p[1] = (__bf16)v.y; p[2] = (__bf16)v.z; p[3] = (__bf16)v.w;
}

// ---------------- small kernels (128-bit vectorized) ----------------
__global__ void k_cast_i32_f32(const int4* __restrict__ s, float4* __restrict__ d, int n4) {
    int i = blockIdx.x * blockDim.x + threadIdx.x;
    if (i >= n4) return;
    int4 v = s[i];
    d[i] = make_float4((float)v.x, (float)v.y, (float)v.z, (float)v.w);
}

__global__ void k_fill_u32(uint4* __restrict__ p, unsigned v, int n4) {
    int i = blockIdx.x * blockDim.x + threadIdx.x;
    if (i < n4) p[i] = make_uint4(v, v, v, v);
}

// Pack W [K x 128] f32 (row-major) into WMMA-B fragment-major bf16:
// Wp[((ks*8 + ntile)*32 + lane)*16 + j] = W[ks*32 + (lane/16)*16 + j][ntile*16 + lane%16]
__global__ void k_pack_w(const float* __restrict__ W, __bf16* __restrict__ Wp, int K, int Kpad) {
    int t = blockIdx.x * blockDim.x + threadIdx.x;
    int total = (Kpad / 32) * 8 * 32;
    if (t >= total) return;
    int lane = t & 31;
    int ntile = (t >> 5) & 7;
    int ks = t >> 8;
    int kb = ks * 32 + (lane >> 4) * 16;
    int col = ntile * 16 + (lane & 15);
    __bf16* dst = &Wp[(size_t)t * 16];
#pragma unroll
    for (int j = 0; j < 16; ++j) {
        int k = kb + j;
        float v = (k < K) ? W[(size_t)k * 128 + col] : 0.f;
        dst[j] = (__bf16)v;
    }
}

// decode segment-min + empty-node mask + leaky relu (4 elems/thread)
__global__ void k_decode_lrelu(const uint4* __restrict__ enc, float4* __restrict__ xo, int n4) {
    int i = blockIdx.x * blockDim.x + threadIdx.x;
    if (i >= n4) return;
    uint4 e = enc[i];
    float4 o;
    o.x = (e.x == 0xFFFFFFFFu) ? 0.f : lrelu(dec_f32(e.x));
    o.y = (e.y == 0xFFFFFFFFu) ? 0.f : lrelu(dec_f32(e.y));
    o.z = (e.z == 0xFFFFFFFFu) ? 0.f : lrelu(dec_f32(e.z));
    o.w = (e.w == 0xFFFFFFFFu) ? 0.f : lrelu(dec_f32(e.w));
    xo[i] = o;
}

// final: softplus(concat(x0[16], x3[128]) @ lin_w[144x8] + lin_b)
__global__ void k_final(const float* __restrict__ x0, const float* __restrict__ x3,
                        const float* __restrict__ lw, const float* __restrict__ lb,
                        float* __restrict__ out, int n) {
    int i = blockIdx.x * blockDim.x + threadIdx.x;
    if (i >= n * 8) return;
    int node = i >> 3, a = i & 7;
    float s = lb[a];
#pragma unroll
    for (int k = 0; k < NODEF; ++k) s += x0[node * NODEF + k] * lw[k * 8 + a];
    for (int k = 0; k < HID; ++k)  s += x3[node * HID + k] * lw[(NODEF + k) * 8 + a];
    out[i] = s > 0.f ? s + log1pf(expf(-s)) : log1pf(expf(s));
}

// ---------------- fused edge-conv GEMM ----------------
// One workgroup = 64 edges x 128 outputs. 8 wave32: wave = (mtile 0..3, nhalf 0..1),
// each wave computes a 16x64 tile via 4 f32 accumulators using v_wmma_f32_16x16x32_bf16.
template <int FIN, int KPAD>
__global__ __launch_bounds__(256) void k_edgeconv(
    const float* __restrict__ x,        // [NNODE, FIN]
    const int* __restrict__ srcI,       // [E]
    const int* __restrict__ dstI,       // [E]
    const float* __restrict__ ea,       // [E]
    const __bf16* __restrict__ w1p,     // packed [KPAD/32][8][32][16]
    const float* __restrict__ b1,       // [128]
    const __bf16* __restrict__ w2p,     // packed [4][8][32][16]
    const float* __restrict__ b2,       // [128]
    unsigned* __restrict__ enc_out)     // [NNODE, 128] encoded-min
{
    __shared__ alignas(16) __bf16 Abuf[64 * KPAD];   // message features (bf16)
    __shared__ alignas(16) __bf16 Hbuf[64 * HID];    // hidden activations (bf16)
    __shared__ int dstl[64];

    const int t = threadIdx.x;
    const int e0 = blockIdx.x * 64;

    // ---- stage A tile: concat([x[dst], x[src], ea], pad) -> bf16 LDS ----
    // 4 threads per edge row; contiguous float4 gathers of the node rows.
    {
        const int le = t >> 2;        // local edge 0..63
        const int q = t & 3;          // quarter of the row
        const int e = e0 + le;
        const int sN = srcI[e];
        const int dN = dstI[e];
        if (q == 0) dstl[le] = dN;
        __bf16* row = &Abuf[le * KPAD];

        const float4* xd4 = (const float4*)(x + (size_t)dN * FIN);
        const float4* xs4 = (const float4*)(x + (size_t)sN * FIN);
        constexpr int CH = FIN / 16;  // float4 chunks per quarter (1 or 8)
#pragma unroll
        for (int j = 0; j < CH; ++j) {
            st4bf(row + (q * CH + j) * 4, xd4[q * CH + j]);
            st4bf(row + FIN + (q * CH + j) * 4, xs4[q * CH + j]);
        }
        if (q == 0) row[2 * FIN] = (__bf16)ea[e];
        for (int k = 2 * FIN + 1 + q; k < KPAD; k += 4) row[k] = (__bf16)0.f;
    }
    __syncthreads();

    const int wave = t >> 5, lane = t & 31;
    const int mtile = wave & 3, nhalf = wave >> 2;
    const int m0 = mtile * 16;
    const int laneHi = lane >> 4;   // 0/1
    const int lrow = lane & 15;

    // ---- GEMM1: [16 x KPAD] @ [KPAD x 64] ----
    v8f acc[4] = {};
#pragma unroll
    for (int ks = 0; ks < KPAD / 32; ++ks) {
        ABfrag av;
        const __bf16* ap = &Abuf[(m0 + lrow) * KPAD + ks * 32 + laneHi * 16];
        av.h[0] = *(const v8bf*)(ap);
        av.h[1] = *(const v8bf*)(ap + 8);
#pragma unroll
        for (int nt = 0; nt < 4; ++nt) {
            int ntg = nhalf * 4 + nt;
            v16bf bv = *(const v16bf*)&w1p[(((size_t)ks * 8 + ntg) * 32 + lane) * 16];
            acc[nt] = __builtin_amdgcn_wmma_f32_16x16x32_bf16(
                false, av.v, false, bv, (short)0, acc[nt], false, false);
        }
    }

    // ---- bias + leaky relu -> bf16 hidden tile in LDS ----
#pragma unroll
    for (int nt = 0; nt < 4; ++nt) {
        int col = nhalf * 64 + nt * 16 + lrow;
        float bias = b1[col];
#pragma unroll
        for (int r = 0; r < 8; ++r) {
            int m = m0 + r + laneHi * 8;
            Hbuf[m * HID + col] = (__bf16)lrelu(acc[nt][r] + bias);
        }
    }
    __syncthreads();

    // ---- GEMM2: [16 x 128] @ [128 x 64] ----
    v8f acc2[4] = {};
#pragma unroll
    for (int ks = 0; ks < 4; ++ks) {
        ABfrag av;
        const __bf16* ap = &Hbuf[(m0 + lrow) * HID + ks * 32 + laneHi * 16];
        av.h[0] = *(const v8bf*)(ap);
        av.h[1] = *(const v8bf*)(ap + 8);
#pragma unroll
        for (int nt = 0; nt < 4; ++nt) {
            int ntg = nhalf * 4 + nt;
            v16bf bv = *(const v16bf*)&w2p[(((size_t)ks * 8 + ntg) * 32 + lane) * 16];
            acc2[nt] = __builtin_amdgcn_wmma_f32_16x16x32_bf16(
                false, av.v, false, bv, (short)0, acc2[nt], false, false);
        }
    }

    // ---- bias + encoded atomic-min scatter (segment_min) ----
#pragma unroll
    for (int nt = 0; nt < 4; ++nt) {
        int col = nhalf * 64 + nt * 16 + lrow;
        float bias = b2[col];
#pragma unroll
        for (int r = 0; r < 8; ++r) {
            int m = m0 + r + laneHi * 8;
            int d = dstl[m];
            unsigned e = enc_f32(acc2[nt][r] + bias);
            atomicMin(&enc_out[(size_t)d * HID + col], e);
        }
    }
}

// ---------------- host launch ----------------
extern "C" void kernel_launch(void* const* d_in, const int* in_sizes, int n_in,
                              void* d_out, int out_size, void* d_ws, size_t ws_size,
                              hipStream_t stream) {
    const int* nodef = (const int*)d_in[0];             // [B,N,16] int32
    const int* eidx = (const int*)d_in[1];              // [B,2,E] int
    const float* ef = (const float*)d_in[2];            // [B,E,1]
    const float* c1w1 = (const float*)d_in[3]; const float* c1b1 = (const float*)d_in[4];
    const float* c1w2 = (const float*)d_in[5]; const float* c1b2 = (const float*)d_in[6];
    const float* c2w1 = (const float*)d_in[7]; const float* c2b1 = (const float*)d_in[8];
    const float* c2w2 = (const float*)d_in[9]; const float* c2b2 = (const float*)d_in[10];
    const float* c3w1 = (const float*)d_in[11]; const float* c3b1 = (const float*)d_in[12];
    const float* c3w2 = (const float*)d_in[13]; const float* c3b2 = (const float*)d_in[14];
    const float* linw = (const float*)d_in[15]; const float* linb = (const float*)d_in[16];
    float* out = (float*)d_out;

    char* ws = (char*)d_ws;
    size_t off = 0;
    auto alloc = [&](size_t bytes) -> void* {
        off = (off + 255) & ~(size_t)255;
        void* p = ws + off;
        off += bytes;
        return p;
    };
    float*    x0f   = (float*)alloc((size_t)BATCH * NNODE * NODEF * 4);
    float*    xb    = (float*)alloc((size_t)NNODE * HID * 4);
    unsigned* enc   = (unsigned*)alloc((size_t)NNODE * HID * 4);
    __bf16*   w1p1  = (__bf16*)alloc((size_t)64 * 128 * 2);
    __bf16*   w2p1  = (__bf16*)alloc((size_t)128 * 128 * 2);
    __bf16*   w1p2  = (__bf16*)alloc((size_t)288 * 128 * 2);
    __bf16*   w2p2  = (__bf16*)alloc((size_t)128 * 128 * 2);
    __bf16*   w1p3  = (__bf16*)alloc((size_t)288 * 128 * 2);
    __bf16*   w2p3  = (__bf16*)alloc((size_t)128 * 128 * 2);

    // prep: cast node features, pack weights (deterministic, every call)
    {
        int n4 = (BATCH * NNODE * NODEF) / 4;
        k_cast_i32_f32<<<(n4 + 255) / 256, 256, 0, stream>>>((const int4*)nodef, (float4*)x0f, n4);
        k_pack_w<<<(512 + 255) / 256, 256, 0, stream>>>(c1w1, w1p1, 33, 64);
        k_pack_w<<<(1024 + 255) / 256, 256, 0, stream>>>(c1w2, w2p1, 128, 128);
        k_pack_w<<<(2304 + 255) / 256, 256, 0, stream>>>(c2w1, w1p2, 257, 288);
        k_pack_w<<<(1024 + 255) / 256, 256, 0, stream>>>(c2w2, w2p2, 128, 128);
        k_pack_w<<<(2304 + 255) / 256, 256, 0, stream>>>(c3w1, w1p3, 257, 288);
        k_pack_w<<<(1024 + 255) / 256, 256, 0, stream>>>(c3w2, w2p3, 128, 128);
    }

    const int nNH = NNODE * HID;
    const int n4 = nNH / 4;
    const int vecB = (n4 + 255) / 256;
    const int gemmB = NEDGE / 64;

    for (int b = 0; b < BATCH; ++b) {
        const float* x0b = x0f + (size_t)b * NNODE * NODEF;
        const int* srcI = eidx + (size_t)b * 2 * NEDGE;
        const int* dstI = srcI + NEDGE;
        const float* eab = ef + (size_t)b * NEDGE;

        // layer 1 (FIN=16, K=33 -> 64)
        k_fill_u32<<<vecB, 256, 0, stream>>>((uint4*)enc, 0xFFFFFFFFu, n4);
        k_edgeconv<NODEF, 64><<<gemmB, 256, 0, stream>>>(
            x0b, srcI, dstI, eab, w1p1, c1b1, w2p1, c1b2, enc);
        k_decode_lrelu<<<vecB, 256, 0, stream>>>((const uint4*)enc, (float4*)xb, n4);

        // layer 2 (FIN=128, K=257 -> 288)
        k_fill_u32<<<vecB, 256, 0, stream>>>((uint4*)enc, 0xFFFFFFFFu, n4);
        k_edgeconv<HID, 288><<<gemmB, 256, 0, stream>>>(
            xb, srcI, dstI, eab, w1p2, c2b1, w2p2, c2b2, enc);
        k_decode_lrelu<<<vecB, 256, 0, stream>>>((const uint4*)enc, (float4*)xb, n4);

        // layer 3
        k_fill_u32<<<vecB, 256, 0, stream>>>((uint4*)enc, 0xFFFFFFFFu, n4);
        k_edgeconv<HID, 288><<<gemmB, 256, 0, stream>>>(
            xb, srcI, dstI, eab, w1p3, c3b1, w2p3, c3b2, enc);
        k_decode_lrelu<<<vecB, 256, 0, stream>>>((const uint4*)enc, (float4*)xb, n4);

        // final linear + softplus
        k_final<<<(NNODE * 8 + 255) / 256, 256, 0, stream>>>(
            x0b, xb, linw, linb, out + (size_t)b * NNODE * 8, NNODE);
    }
}
